// LowRankChristoffel_12592844112397
// MI455X (gfx1250) — matrix operand order repaired and checked by reading, not compiled, not run on hardware
//
#include <hip/hip_runtime.h>
#include <stdint.h>

// Problem constants (fixed by the reference)
#define DIM    256
#define RANK   16
#define BATCH  262144
#define BM     64            // batch rows per block
#define KCAT   512           // concat K: [x | force] vs [Wf | Wi]
#define NT_J   16            // 16 column tiles of 16 over DIM
#define NT_K   16            // 16 k-steps of 32 over KCAT
#define XF_STRIDE 520        // 512 + 8 bf16 pad -> conflict-free ds_load_b128
#define V_STRIDE  264        // 256 + 8 bf16 pad

#define FRICTION_SCALE 0.1f
#define VEL_FR         0.1f
#define EPS            1e-6f

typedef __attribute__((ext_vector_type(16))) __bf16 v16bf;
typedef __attribute__((ext_vector_type(8)))  __bf16 v8bf;
typedef __attribute__((ext_vector_type(4)))  __bf16 v4bf;
typedef __attribute__((ext_vector_type(8)))  float  v8f;
typedef __attribute__((ext_vector_type(4)))  float  v4f;

__device__ __forceinline__ uint16_t f2bf(float f) {
  uint32_t u = __float_as_uint(f);
  u += 0x7FFFu + ((u >> 16) & 1u);   // round-to-nearest-even
  return (uint16_t)(u >> 16);
}

// ---------------------------------------------------------------------------
// Prep: pack [Wf|Wi] (K=512) and U into bf16 B-fragment layout for
// v_wmma_f32_16x16x32_bf16.  B (32x16, 16-bit): lane l -> column l&15,
// element e (0..15 within lane) -> K = (l>=16 ? 16 : 0) + e (this code's order; A matches).
// Fragment tile = 32 lanes x 16 bf16 = 1024 B, contiguous per (kt, jt).
// ---------------------------------------------------------------------------
__global__ void prep_weights(const float* __restrict__ U,
                             const float* __restrict__ Wf,
                             const float* __restrict__ Wi,
                             uint16_t* __restrict__ bfrag,   // [NT_K][NT_J][32][16]
                             uint16_t* __restrict__ ufrag)   // [8][32][16]
{
  const int TOTB = NT_K * NT_J * 32 * 16;   // 131072
  const int TOTU = 8 * 32 * 16;             // 4096
  for (int f = blockIdx.x * blockDim.x + threadIdx.x; f < TOTB + TOTU;
       f += gridDim.x * blockDim.x) {
    if (f < TOTB) {
      int kt   = f / (NT_J * 32 * 16);
      int rem  = f % (NT_J * 32 * 16);
      int jt   = rem / (32 * 16);
      int rem2 = rem % (32 * 16);
      int lane = rem2 / 16;
      int e    = rem2 % 16;
      int k = kt * 32 + ((lane >> 4) * 16) + e;          // 0..511
      int j = jt * 16 + (lane & 15);                      // 0..255
      float val = (k < DIM) ? Wf[(size_t)j * DIM + k]
                            : Wi[(size_t)j * DIM + (k - DIM)];
      bfrag[f] = f2bf(val);
    } else {
      int g    = f - TOTB;
      int kt   = g / (32 * 16);
      int rem2 = g % (32 * 16);
      int lane = rem2 / 16;
      int e    = rem2 % 16;
      int k = kt * 32 + ((lane >> 4) * 16) + e;          // 0..255
      int n = lane & 15;                                  // rank 0..15
      ufrag[g] = f2bf(U[(size_t)k * RANK + n]);
    }
  }
}

// ---------------------------------------------------------------------------
// Main fused kernel: 256 threads = 8 waves; wave w -> row tile (w&3),
// column half (w>>2).  Each wave: 16 rows x 128 cols of output.
// ---------------------------------------------------------------------------
__global__ void __launch_bounds__(256)
lrchristoffel_main(const float* __restrict__ v,
                   const float* __restrict__ x,
                   const float* __restrict__ force,
                   const float* __restrict__ W,      // [DIM][RANK]
                   const float* __restrict__ bfv,    // [DIM]
                   const uint16_t* __restrict__ bfrag,
                   const uint16_t* __restrict__ ufrag,
                   float* __restrict__ out)
{
  extern __shared__ uint8_t smem[];
  uint16_t* xf    = (uint16_t*)smem;                                   // BM x XF_STRIDE bf16
  uint16_t* vl    = (uint16_t*)(smem + BM * XF_STRIDE * 2);            // BM x V_STRIDE bf16
  float*    vnorm = (float*)   (smem + BM * XF_STRIDE * 2 + BM * V_STRIDE * 2);

  const int tid     = threadIdx.x;
  const int rowBase = blockIdx.x * BM;

  if (tid < BM) vnorm[tid] = 0.f;
  __syncthreads();

  // ---- stage x|force and v to LDS as bf16 (hw cvt + b64 stores);
  //      accumulate per-row ||v||^2 ----
  {
    int r  = tid >> 2;                 // 0..63
    int cb = (tid & 3) * 64;           // column chunk base
    const v4f* xg = (const v4f*)(x     + (size_t)(rowBase + r) * DIM + cb);
    const v4f* fg = (const v4f*)(force + (size_t)(rowBase + r) * DIM + cb);
    const v4f* vg = (const v4f*)(v     + (size_t)(rowBase + r) * DIM + cb);
    v4bf* xrow = (v4bf*)(xf + r * XF_STRIDE + cb);
    v4bf* frow = (v4bf*)(xf + r * XF_STRIDE + DIM + cb);
    v4bf* vrow = (v4bf*)(vl + r * V_STRIDE + cb);
    float ss = 0.f;
#pragma unroll
    for (int i = 0; i < 16; ++i) {
      v4f a = xg[i];
      xrow[i] = __builtin_convertvector(a, v4bf);
      v4f b = fg[i];
      frow[i] = __builtin_convertvector(b, v4bf);
      v4f c = vg[i];
      vrow[i] = __builtin_convertvector(c, v4bf);
      ss += c.x * c.x + c.y * c.y + c.z * c.z + c.w * c.w;
    }
    atomicAdd(&vnorm[r], ss);          // ds_add_f32
  }
  __syncthreads();

  const int lane   = tid & 31;
  const int wave   = tid >> 5;
  const int mt     = wave & 3;         // row tile within block
  const int ch     = wave >> 2;        // column half (0: cols 0..127, 1: 128..255)
  const int laneHi = lane >> 4;
  const int nl     = lane & 15;
  const int m0l    = mt * 16;
  const int arow   = m0l + nl;         // A-fragment row owned by this lane

  // ---- proj = v @ U  (one 16x16 tile, K=256 -> 8 WMMAs) ----
  v8f pacc{};
#pragma unroll
  for (int kt = 0; kt < 8; ++kt) {
    const uint16_t* ab = vl + arow * V_STRIDE + kt * 32 + laneHi * 16;
    v8bf alo = *(const v8bf*)ab;
    v8bf ahi = *(const v8bf*)(ab + 8);
    v16bf a = __builtin_shufflevector(alo, ahi, 0,1,2,3,4,5,6,7,8,9,10,11,12,13,14,15);
    v16bf b = *(const v16bf*)(ufrag + kt * 512 + lane * 16);
    pacc = __builtin_amdgcn_wmma_f32_16x16x32_bf16(false, a, false, b,
                                                   (short)0, pacc, false, false);
  }

  // ---- sq = proj^2 / (1 + ||proj||)  (row norm via 16-lane xor reduce) ----
  float sqv[8];
#pragma unroll
  for (int q = 0; q < 8; ++q) {
    float p = pacc[q];
    float s = p * p;
    s += __shfl_xor(s, 1, 32);
    s += __shfl_xor(s, 2, 32);
    s += __shfl_xor(s, 4, 32);
    s += __shfl_xor(s, 8, 32);
    float scale = 1.f / (1.f + sqrtf(s) + EPS);
    sqv[q] = p * p * scale;
  }

  // ---- gate = [x|force] @ [Wf|Wi]^T : K=512, 8 col tiles per wave ----
  v8f acc[8];
#pragma unroll
  for (int jt = 0; jt < 8; ++jt) acc[jt] = v8f{};

  for (int kt = 0; kt < NT_K; ++kt) {
    const uint16_t* ab = xf + arow * XF_STRIDE + kt * 32 + laneHi * 16;
    v8bf alo = *(const v8bf*)ab;
    v8bf ahi = *(const v8bf*)(ab + 8);
    v16bf a = __builtin_shufflevector(alo, ahi, 0,1,2,3,4,5,6,7,8,9,10,11,12,13,14,15);
    const uint16_t* bbase = bfrag + ((size_t)(kt * NT_J + ch * 8) * 512) + lane * 16;
    __builtin_prefetch(bbase + (size_t)NT_J * 512, 0, 1);  // next kt's tiles
#pragma unroll
    for (int jt = 0; jt < 8; ++jt) {
      v16bf b = *(const v16bf*)(bbase + (size_t)jt * 512);
      acc[jt] = __builtin_amdgcn_wmma_f32_16x16x32_bf16(false, a, false, b,
                                                        (short)0, acc[jt], false, false);
    }
  }

  // ---- per-row velocity-friction factor (hoisted: 8 sqrt per wave) ----
  float velf[8];
#pragma unroll
  for (int q = 0; q < 8; ++q) {
    float sv = vnorm[m0l + q + laneHi * 8];
    velf[q] = 1.f + VEL_FR * sqrtf(sv) / (16.f + EPS);     // sqrt(DIM)=16
  }

  // ---- epilogue: gamma = sq@W^T + mu*v ; out = 10*tanh(gamma/10) ----
#pragma unroll
  for (int jt = 0; jt < 8; ++jt) {
    const int jtG = ch * 8 + jt;
    const int j   = jtG * 16 + nl;            // output column of this lane

    // rank-16 contraction: broadcast sq[M][rr] from lane rr of own half
    float glr[8] = {0,0,0,0,0,0,0,0};
    const float* wrow = W + (size_t)j * RANK;
#pragma unroll
    for (int rr = 0; rr < RANK; ++rr) {
      float wvv = wrow[rr];
      int src = (lane & 16) | rr;
#pragma unroll
      for (int q = 0; q < 8; ++q)
        glr[q] = fmaf(__shfl(sqv[q], src, 32), wvv, glr[q]);
    }

    float bias = bfv[j];
#pragma unroll
    for (int q = 0; q < 8; ++q) {
      int rg = rowBase + m0l + q + laneHi * 8;             // global row
      float g  = acc[jt][q] + bias;
      float mu = (FRICTION_SCALE / (1.f + __expf(-g))) * velf[q];
      float vv = v[(size_t)rg * DIM + j];
      float t  = glr[q] + mu * vv;
      out[(size_t)rg * DIM + j] = 10.f * tanhf(t * 0.1f);
    }
  }
}

// ---------------------------------------------------------------------------
extern "C" void kernel_launch(void* const* d_in, const int* in_sizes, int n_in,
                              void* d_out, int out_size, void* d_ws, size_t ws_size,
                              hipStream_t stream) {
  const float* v     = (const float*)d_in[0];
  const float* x     = (const float*)d_in[1];
  const float* force = (const float*)d_in[2];
  const float* U     = (const float*)d_in[3];
  const float* W     = (const float*)d_in[4];
  const float* Wf    = (const float*)d_in[5];
  const float* bfv   = (const float*)d_in[6];
  const float* Wi    = (const float*)d_in[7];

  uint16_t* bfrag = (uint16_t*)d_ws;                       // 256 KB
  uint16_t* ufrag = bfrag + (size_t)NT_K * NT_J * 512;     //   8 KB

  prep_weights<<<128, 256, 0, stream>>>(U, Wf, Wi, bfrag, ufrag);

  size_t lds = (size_t)BM * XF_STRIDE * 2 + (size_t)BM * V_STRIDE * 2 + BM * 4;
  lrchristoffel_main<<<BATCH / BM, 256, lds, stream>>>(
      v, x, force, W, bfv, bfrag, ufrag, (float*)d_out);
}
